// CustomMultiHeadAttention_4810363372438
// MI455X (gfx1250) — matrix operand order repaired and checked by reading, not compiled, not run on hardware
//
#include <hip/hip_runtime.h>
#include <hip/hip_bf16.h>
#include <stdint.h>

// ---------------------------------------------------------------------------
// MHA + RoPE for MI455X (gfx1250, wave32, WMMA f32_16x16x32_f16,
// async global->LDS double-buffered flash attention)
//   B=2, S=2048, H=16, D=64, DM=1024
// ---------------------------------------------------------------------------

#define NUM_HEADS 16
#define KEY_DIM   64
#define D_MODEL   1024
#define BATCH     2
#define SEQ       2048

typedef __attribute__((ext_vector_type(16))) _Float16 v16h;
typedef __attribute__((ext_vector_type(8)))  _Float16 v8h;
typedef __attribute__((ext_vector_type(8)))  float    v8f;

// ---- WMMA helper: D = A(16x32 f16) * B(32x16 f16) + C(16x16 f32) ----------
__device__ __forceinline__ v8f wmma16(v16h a, v16h b, v8f c) {
  return __builtin_amdgcn_wmma_f32_16x16x32_f16(
      /*neg_a=*/false, a, /*neg_b=*/false, b,
      /*c_mod=*/(short)0, c, /*reuse_a=*/false, /*reuse_b=*/false);
}

// ---- A-fragment load (16x32 f16 layout): lane m=l%16, half=l/16.
// element i (0..7)  -> k = 8*half + i ; element i (8..15) -> k = 16+8*half+i-8
__device__ __forceinline__ v16h frag_a(const _Float16* p, int half) {
  v8h lo = *(const v8h*)(p + 8 * half);
  v8h hi = *(const v8h*)(p + 16 + 8 * half);
  return __builtin_shufflevector(lo, hi, 0, 1, 2, 3, 4, 5, 6, 7,
                                 8, 9, 10, 11, 12, 13, 14, 15);
}

// ---- B-fragment load: lane n=l%16, half=l/16; element i (0..7) -> k = 8*half + i ; element i (8..15) -> k = 16+8*half+i-8
__device__ __forceinline__ v16h frag_b(const _Float16* p, int half) {
  v8h lo = *(const v8h*)(p + 8 * half);
  v8h hi = *(const v8h*)(p + 16 + 8 * half);
  return __builtin_shufflevector(lo, hi, 0, 1, 2, 3, 4, 5, 6, 7,
                                 8, 9, 10, 11, 12, 13, 14, 15);
}

// ---- CDNA5 async global->LDS 16-byte copy (ASYNCcnt tracked) --------------
__device__ __forceinline__ void async_cp16(uint32_t lds_addr, const void* g) {
  asm volatile("global_load_async_to_lds_b128 %0, %1, off"
               :
               : "v"(lds_addr), "v"((uint64_t)(uintptr_t)g)
               : "memory");
}
__device__ __forceinline__ void wait_async0() {
  asm volatile("s_wait_asynccnt 0x0" ::: "memory");
}
__device__ __forceinline__ void wait_ds0() {
  asm volatile("s_wait_dscnt 0x0" ::: "memory");
}

// ===========================================================================
// Tiled GEMM: C[4096 x 1024] = A[4096 x 1024] * W[1024 x 1024] + bias
//   block = 128 threads (4 waves), tile 64x64, K-step 32.
// OUTMODE 0: f32 out, token-major (final projection)
// OUTMODE 1: f16 out, head-major [B*H][S][64], bias only          (V)
// OUTMODE 2: f16 out, head-major, bias + RoPE                     (Q, K)
// ===========================================================================
template <bool A_F16, int OUTMODE>
__global__ __launch_bounds__(128) void gemm_kernel(
    const void* __restrict__ Aptr, const float* __restrict__ W,
    const float* __restrict__ bias, void* __restrict__ Out) {
  __shared__ __align__(32) _Float16 lds_a[64 * 32];   // [m][k]
  __shared__ __align__(32) _Float16 lds_bT[64 * 32];  // [n][k]

  const int tid = threadIdx.x;
  const int lane = tid & 31;
  const int wave = tid >> 5;
  const int lanelo = lane & 15;
  const int half = lane >> 4;
  const int rowBase = blockIdx.x * 64;
  const int colBase = blockIdx.y * 64;

  v8f acc[4] = {};

  for (int kt = 0; kt < D_MODEL / 32; ++kt) {
    __syncthreads();
    const int kBase = kt * 32;
#pragma unroll
    for (int j = 0; j < 16; ++j) {
      int idx = tid + 128 * j;  // 0..2047, row-major [64][32]
      int m = idx >> 5, kk = idx & 31;
      size_t g = (size_t)(rowBase + m) * D_MODEL + kBase + kk;
      _Float16 hv;
      if (A_F16)
        hv = ((const _Float16*)Aptr)[g];
      else
        hv = (_Float16)(((const float*)Aptr)[g]);
      lds_a[idx] = hv;
    }
#pragma unroll
    for (int j = 0; j < 16; ++j) {
      int idx = tid + 128 * j;  // kk*64 + n
      int kk = idx >> 6, n = idx & 63;
      lds_bT[n * 32 + kk] =
          (_Float16)W[(size_t)(kBase + kk) * D_MODEL + colBase + n];
    }
    __syncthreads();

    const _Float16* arow = lds_a + (wave * 16 + lanelo) * 32;
    v16h af = frag_a(arow, half);
#pragma unroll
    for (int t = 0; t < 4; ++t) {
      v16h bf = frag_b(lds_bT + (16 * t + lanelo) * 32, half);
      acc[t] = wmma16(af, bf, acc[t]);
    }
  }

  // -------- epilogue --------
  const int mBase = rowBase + wave * 16;
  float bvals[4];
#pragma unroll
  for (int t = 0; t < 4; ++t) bvals[t] = bias[colBase + 16 * t + lanelo];

  if (OUTMODE == 0) {
    float* out = (float*)Out;
#pragma unroll
    for (int r = 0; r < 8; ++r) {
      int row = mBase + r + 8 * half;
#pragma unroll
      for (int t = 0; t < 4; ++t)
        out[(size_t)row * D_MODEL + colBase + 16 * t + lanelo] =
            acc[t][r] + bvals[t];
    }
  } else {
    _Float16* out = (_Float16*)Out;  // [B*H][S][64]
    const int h = colBase >> 6;
#pragma unroll
    for (int r = 0; r < 8; ++r) {
      int row = mBase + r + 8 * half;
      int bb = row / SEQ, ss = row % SEQ;
      size_t obase = ((size_t)(bb * NUM_HEADS + h) * SEQ + ss) * KEY_DIM;
      if (OUTMODE == 1) {
#pragma unroll
        for (int t = 0; t < 4; ++t)
          out[obase + 16 * t + lanelo] = (_Float16)(acc[t][r] + bvals[t]);
      } else {
        // RoPE: pair (d, d+32) == accumulator tiles (t, t+2), same lane/vgpr
#pragma unroll
        for (int t = 0; t < 2; ++t) {
          int d = 16 * t + lanelo;  // 0..31
          float x1 = acc[t][r] + bvals[t];
          float x2 = acc[t + 2][r] + bvals[t + 2];
          float fr = __expf(-(float)d * 0.2878231366242557f);  // 10000^(-d/32)
          float ang = (float)ss * fr;
          float sn, cs;
          __sincosf(ang, &sn, &cs);
          out[obase + d] = (_Float16)(cs * x1 - sn * x2);
          out[obase + d + 32] = (_Float16)(sn * x1 + cs * x2);
        }
      }
    }
  }
}

// ===========================================================================
// One-shot V transpose: [B*H][S][64] -> [B*H][64][S]  (coalesced both sides)
// so attention can stage V^T with verbatim async 16B copies.
// ===========================================================================
__global__ __launch_bounds__(128) void transpose_v_kernel(
    const _Float16* __restrict__ Vh, _Float16* __restrict__ Vt) {
  __shared__ _Float16 t[64][65];  // padded to dodge bank conflicts
  const int tid = threadIdx.x;
  const int sBase = blockIdx.x * 64;
  const int bh = blockIdx.y;

  const uint32_t* src =
      (const uint32_t*)(Vh + ((size_t)bh * SEQ + sBase) * KEY_DIM);
#pragma unroll
  for (int j = 0; j < 16; ++j) {
    int idx = tid + 128 * j;  // dwords over [64 s][32 dpairs]
    int s = idx >> 5, p = idx & 31;
    union { uint32_t u; _Float16 h[2]; } cv;
    cv.u = src[(size_t)s * 32 + p];
    t[2 * p][s] = cv.h[0];
    t[2 * p + 1][s] = cv.h[1];
  }
  __syncthreads();
  uint32_t* dst = (uint32_t*)(Vt + (size_t)bh * KEY_DIM * SEQ);
#pragma unroll
  for (int j = 0; j < 16; ++j) {
    int idx = tid + 128 * j;  // dwords over [64 d][32 spairs]
    int d = idx >> 5, sp = idx & 31;
    union { uint32_t u; _Float16 h[2]; } cv;
    cv.h[0] = t[d][2 * sp];
    cv.h[1] = t[d][2 * sp + 1];
    dst[(size_t)d * (SEQ / 2) + (sBase >> 1) + sp] = cv.u;
  }
}

// ===========================================================================
// Flash attention per (b,h): grid.x = S/64 query blocks, grid.y = B*H.
// 128 threads = 4 waves; each wave owns 16 query rows.
// K and V^T tiles double-buffered via GLOBAL_LOAD_ASYNC_TO_LDS_B128.
// ===========================================================================
__global__ __launch_bounds__(128) void attn_kernel(
    const _Float16* __restrict__ Qh, const _Float16* __restrict__ Kh,
    const _Float16* __restrict__ Vt, _Float16* __restrict__ Ctx) {
  __shared__ __align__(32) _Float16 lds_k[2][64 * 64];   // [key][d]
  __shared__ __align__(32) _Float16 lds_vT[2][64 * 64];  // [d][key]
  __shared__ __align__(32) _Float16 lds_p[4 * 16 * 64];  // Q stage, then P

  const int tid = threadIdx.x;
  const int lane = tid & 31;
  const int wave = tid >> 5;
  const int lanelo = lane & 15;
  const int half = lane >> 4;
  const int qBase = blockIdx.x * 64;
  const int bh = blockIdx.y;

  const char* Qbp = (const char*)(Qh + (size_t)bh * SEQ * KEY_DIM);
  const char* Kbp = (const char*)(Kh + (size_t)bh * SEQ * KEY_DIM);
  const char* Vtp = (const char*)(Vt + (size_t)bh * KEY_DIM * SEQ);

  const uint32_t ldsK = (uint32_t)(uintptr_t)&lds_k[0][0];
  const uint32_t ldsV = (uint32_t)(uintptr_t)&lds_vT[0][0];
  const uint32_t ldsP = (uint32_t)(uintptr_t)&lds_p[0];

  // ---- async-stage Q block into lds_p region (8 KB, verbatim) ----
#pragma unroll
  for (int j = 0; j < 4; ++j) {
    int c = tid + 128 * j;  // 512 x 16B chunks
    async_cp16(ldsP + c * 16, Qbp + (size_t)qBase * 128 + (size_t)c * 16);
  }
  wait_async0();
  __syncthreads();

  v16h qf[2];
  {
    const _Float16* qrow = lds_p + (wave * 16 + lanelo) * 64;
    qf[0] = frag_a(qrow, half);
    qf[1] = frag_a(qrow + 32, half);
  }

  // ---- issue stage 0 (K row-major verbatim; V^T verbatim rows) ----
  auto stage = [&](int buf, int keyBase) {
#pragma unroll
    for (int j = 0; j < 4; ++j) {
      int c = tid + 128 * j;  // K block: 512 x 16B, contiguous
      async_cp16(ldsK + buf * 8192 + c * 16,
                 Kbp + (size_t)keyBase * 128 + (size_t)c * 16);
    }
#pragma unroll
    for (int j = 0; j < 4; ++j) {
      int c = tid + 128 * j;  // V^T block: 64 d-rows x 128B
      int d = c >> 3, off = (c & 7) * 16;
      async_cp16(ldsV + buf * 8192 + d * 128 + off,
                 Vtp + ((size_t)d * SEQ + keyBase) * 2 + off);
    }
  };
  stage(0, 0);

  float rmax[8], rsum[8];
  v8f cacc[4] = {};
#pragma unroll
  for (int r = 0; r < 8; ++r) { rmax[r] = -1e30f; rsum[r] = 0.0f; }

  const float scale = 0.125f;  // 1/sqrt(64)

  for (int kb = 0; kb < SEQ / 64; ++kb) {
    wait_async0();    // this wave's stage-kb chunks landed in LDS
    __syncthreads();  // everyone's chunks landed; prior readers done
    if (kb + 1 < SEQ / 64) stage((kb + 1) & 1, (kb + 1) * 64);

    const _Float16* ck = lds_k[kb & 1];
    const _Float16* cv = lds_vT[kb & 1];

    // ---- scores S = Q*K^T (16x64 per wave) ----
    v8f sacc[4] = {};
#pragma unroll
    for (int ks = 0; ks < 2; ++ks)
#pragma unroll
      for (int t = 0; t < 4; ++t) {
        v16h bf = frag_b(ck + (16 * t + lanelo) * 64 + 32 * ks, half);
        sacc[t] = wmma16(qf[ks], bf, sacc[t]);
      }

    // ---- online softmax (row m = r + 8*half; keys across 16-lane group) ----
    _Float16* pb = lds_p + wave * 16 * 64;
#pragma unroll
    for (int r = 0; r < 8; ++r) {
      float s0 = sacc[0][r] * scale, s1 = sacc[1][r] * scale;
      float s2 = sacc[2][r] * scale, s3 = sacc[3][r] * scale;
      float mx = fmaxf(fmaxf(s0, s1), fmaxf(s2, s3));
#pragma unroll
      for (int off = 1; off < 16; off <<= 1)
        mx = fmaxf(mx, __shfl_xor(mx, off, 32));
      float nm = fmaxf(rmax[r], mx);
      float corr = __expf(rmax[r] - nm);
      rmax[r] = nm;
      float p0 = __expf(s0 - nm), p1 = __expf(s1 - nm);
      float p2 = __expf(s2 - nm), p3 = __expf(s3 - nm);
      float ps = p0 + p1 + p2 + p3;
#pragma unroll
      for (int off = 1; off < 16; off <<= 1) ps += __shfl_xor(ps, off, 32);
      rsum[r] = rsum[r] * corr + ps;
      int prow = (r + 8 * half) * 64 + lanelo;
      pb[prow] = (_Float16)p0;
      pb[prow + 16] = (_Float16)p1;
      pb[prow + 32] = (_Float16)p2;
      pb[prow + 48] = (_Float16)p3;
#pragma unroll
      for (int t = 0; t < 4; ++t) cacc[t][r] *= corr;
    }
    wait_ds0();  // wave-local C-layout -> A-layout relayout through LDS

    // ---- ctx += P*V ----
#pragma unroll
    for (int ks = 0; ks < 2; ++ks) {
      v16h pf = frag_a(pb + lanelo * 64 + 32 * ks, half);
#pragma unroll
      for (int t = 0; t < 4; ++t) {
        v16h vf = frag_b(cv + (16 * t + lanelo) * 64 + 32 * ks, half);
        cacc[t] = wmma16(pf, vf, cacc[t]);
      }
    }
  }

  // ---- normalize and store ctx token-major f16 [B][S][1024] ----
  const int b = bh >> 4, h = bh & 15;
#pragma unroll
  for (int r = 0; r < 8; ++r) {
    int srow = qBase + wave * 16 + r + 8 * half;
    float inv = 1.0f / rsum[r];
    size_t obase = ((size_t)(b * SEQ + srow)) * D_MODEL + h * KEY_DIM;
#pragma unroll
    for (int t = 0; t < 4; ++t)
      Ctx[obase + 16 * t + lanelo] = (_Float16)(cacc[t][r] * inv);
  }
}

// ===========================================================================
extern "C" void kernel_launch(void* const* d_in, const int* in_sizes, int n_in,
                              void* d_out, int out_size, void* d_ws,
                              size_t ws_size, hipStream_t stream) {
  const float* x  = (const float*)d_in[0];
  const float* Wq = (const float*)d_in[1];
  const float* bq = (const float*)d_in[2];
  const float* Wk = (const float*)d_in[3];
  const float* bk = (const float*)d_in[4];
  const float* Wv = (const float*)d_in[5];
  const float* bv = (const float*)d_in[6];
  const float* Wo = (const float*)d_in[7];
  const float* bo = (const float*)d_in[8];

  const size_t NTOK = (size_t)BATCH * SEQ;  // 4096
  const size_t ELEM = NTOK * D_MODEL;       // 4M halves per buffer
  _Float16* Qh  = (_Float16*)d_ws;
  _Float16* Kh  = Qh + ELEM;
  _Float16* Vh  = Kh + ELEM;
  _Float16* Vt  = Vh + ELEM;
  _Float16* Ctx = Vt + ELEM;

  dim3 blk(128);
  dim3 gGemm((unsigned)(NTOK / 64), D_MODEL / 64);  // 64 x 16 tiles

  gemm_kernel<false, 2><<<gGemm, blk, 0, stream>>>(x, Wq, bq, Qh);
  gemm_kernel<false, 2><<<gGemm, blk, 0, stream>>>(x, Wk, bk, Kh);
  gemm_kernel<false, 1><<<gGemm, blk, 0, stream>>>(x, Wv, bv, Vh);

  dim3 gTr(SEQ / 64, BATCH * NUM_HEADS);
  transpose_v_kernel<<<gTr, blk, 0, stream>>>(Vh, Vt);

  dim3 gAttn(SEQ / 64, BATCH * NUM_HEADS);  // 32 x 32
  attn_kernel<<<gAttn, blk, 0, stream>>>(Qh, Kh, Vt, Ctx);

  gemm_kernel<true, 0><<<gGemm, blk, 0, stream>>>(Ctx, Wo, bo, d_out);
}
